// SupervisedContrastiveLoss_55336358641764
// MI455X (gfx1250) — matrix operand order repaired and checked by reading, not compiled, not run on hardware
//
#include <hip/hip_runtime.h>

typedef _Float16 half_t;
typedef __attribute__((ext_vector_type(16))) _Float16 v16h;
typedef __attribute__((ext_vector_type(8)))  _Float16 v8h;
typedef __attribute__((ext_vector_type(4)))  _Float16 v4h;
typedef __attribute__((ext_vector_type(8)))  float    v8f;
typedef __attribute__((ext_vector_type(4)))  float    v4f;

#define NROWS 8192
#define KDIM  128
#define EXP_SELF 22026.465794806718f      /* exp(1/tau) = exp(10)            */
#define EXP2_SCALE 14.426950408889634f    /* (1/tau) * log2(e)               */
#define SQRT_SCALE 3.7982825656899547f    /* sqrt((1/tau)*log2(e)); folded   */
                                          /* into both GEMM operands so the */
                                          /* WMMA output feeds exp2 directly */

#define WMMA16(A, B, C) \
  __builtin_amdgcn_wmma_f32_16x16x32_f16(false, (A), false, (B), (short)0, (C), false, false)

// ---------------------------------------------------------------------------
// Kernel 1: L2-normalize rows, pre-scale by sqrt((1/tau)*log2e) -> f16
// workspace; zero per-row accumulators. One wave per row; each lane handles
// 4 of the 128 columns.
// ---------------------------------------------------------------------------
__global__ __launch_bounds__(256) void k_normalize(
    const float* __restrict__ feat, half_t* __restrict__ fnorm,
    float* __restrict__ rowAll, float* __restrict__ rowPos) {
  const int wave = threadIdx.x >> 5;
  const int lane = threadIdx.x & 31;
  const int row  = blockIdx.x * 8 + wave;

  v4f v = *(const v4f*)(feat + (size_t)row * KDIM + lane * 4);
  float s = v.x * v.x + v.y * v.y + v.z * v.z + v.w * v.w;
#pragma unroll
  for (int m = 16; m >= 1; m >>= 1) s += __shfl_xor(s, m, 32);

  const float scale = SQRT_SCALE / fmaxf(sqrtf(s), 1e-12f);
  v4h o;
  o.x = (half_t)(v.x * scale);
  o.y = (half_t)(v.y * scale);
  o.z = (half_t)(v.z * scale);
  o.w = (half_t)(v.w * scale);
  *(v4h*)(fnorm + (size_t)row * KDIM + lane * 4) = o;

  if (lane == 0) { rowAll[row] = 0.0f; rowPos[row] = 0.0f; }
}

// ---------------------------------------------------------------------------
// Kernel 2: fused exp2(scaled f·fᵀ) row reductions via WMMA f16 -> f32.
//   Block: 32 rows (2 row tiles) x 1024-col chunk; 8 waves, each wave owns
//   both row tiles and 8 column tiles. B fragments are shared by the two
//   WMMA accumulator chains -> 8 WMMAs per 8 b128 loads; next-tile B is
//   fetched before the current tile is consumed (1-deep software pipeline).
//   unroll capped at 2 to stay under 256 VGPRs.
// Fragment layouts used here (C/D as in CDNA5 ISA 7.12.2; A/B in contiguous K order):
//   A 16x32 f16 : lane L<16 -> M=L, K=kb..kb+15; lane L+16 ->
//                 M=L, K=kb+16..kb+31.
//   B 32x16 f16 : lane n -> N=n&15, 16 contiguous K from kb + (n>=16 ? 16:0).
//   C/D 16x16   : VGPR r, lanes 0-15 -> D[r][lane]; lanes 16-31 -> D[r+8][..].
// ---------------------------------------------------------------------------
__global__ __launch_bounds__(256) void k_cosexp(
    const half_t* __restrict__ fnorm, const int* __restrict__ labels,
    float* __restrict__ rowAll, float* __restrict__ rowPos) {
  __shared__ float sAll[32];
  __shared__ float sPos[32];

  const int tid  = threadIdx.x;
  const int wave = tid >> 5;
  const int lane = tid & 31;
  const int lm   = lane & 15;   // lane within half
  const int hi   = lane >> 4;   // which 16-lane half
  const int row0 = blockIdx.x * 32;   // two row tiles: row0, row0+16

  if (tid < 32) { sAll[tid] = 0.0f; sPos[tid] = 0.0f; }
  __syncthreads();

  // ---- A fragments for both row tiles (reused across all column tiles) ----
  const half_t* ar0 = fnorm + (size_t)(row0 + lm) * KDIM;
  const half_t* ar1 = ar0 + (size_t)16 * KDIM;
  v16h a0[4], a1[4];
#pragma unroll
  for (int kc = 0; kc < 4; ++kc) {
    const int kb = kc * 32 + hi * 16;
    v8h l0 = *(const v8h*)(ar0 + kb);
    v8h u0 = *(const v8h*)(ar0 + kb + 8);
    v8h l1 = *(const v8h*)(ar1 + kb);
    v8h u1 = *(const v8h*)(ar1 + kb + 8);
#pragma unroll
    for (int i = 0; i < 8; ++i) {
      a0[kc][i] = l0[i]; a0[kc][i + 8] = u0[i];
      a1[kc][i] = l1[i]; a1[kc][i + 8] = u1[i];
    }
  }

  // ---- labels of the 16 output rows this lane owns (8 per row tile) ----
  int rlab0[8], rlab1[8];
#pragma unroll
  for (int r = 0; r < 8; ++r) {
    rlab0[r] = labels[row0 + hi * 8 + r];
    rlab1[r] = labels[row0 + 16 + hi * 8 + r];
  }

  float aAll0[8], aPos0[8], aAll1[8], aPos1[8];
#pragma unroll
  for (int r = 0; r < 8; ++r) {
    aAll0[r] = 0.0f; aPos0[r] = 0.0f; aAll1[r] = 0.0f; aPos1[r] = 0.0f;
  }

  const int ct0 = blockIdx.y * 64 + wave * 8;  // first column tile for wave

  // ---- prologue: fetch tile 0 B fragments + column label ----
  const half_t* bp = fnorm + (size_t)(ct0 * 16 + lm) * KDIM + hi * 16;
  v16h b0 = *(const v16h*)(bp);
  v16h b1 = *(const v16h*)(bp + 32);
  v16h b2 = *(const v16h*)(bp + 64);
  v16h b3 = *(const v16h*)(bp + 96);
  int clab = labels[ct0 * 16 + lm];

#pragma unroll 2
  for (int t = 0; t < 8; ++t) {
    // ---- issue next tile's loads before consuming the current tile ----
    v16h nb0 = b0, nb1 = b1, nb2 = b2, nb3 = b3;
    int nclab = clab;
    if (t < 7) {
      const half_t* np = fnorm + (size_t)((ct0 + t + 1) * 16 + lm) * KDIM + hi * 16;
      nb0 = *(const v16h*)(np);
      nb1 = *(const v16h*)(np + 32);
      nb2 = *(const v16h*)(np + 64);
      nb3 = *(const v16h*)(np + 96);
      nclab = labels[(ct0 + t + 1) * 16 + lm];
      if (t < 6)  // hint the tile after that into L2/L0
        __builtin_prefetch(fnorm + (size_t)((ct0 + t + 2) * 16 + lm) * KDIM, 0, 1);
    }

    // ---- two independent WMMA chains share the B fragments ----
    v8f c0 = {}, c1 = {};
    c0 = WMMA16(a0[0], b0, c0);  c1 = WMMA16(a1[0], b0, c1);
    c0 = WMMA16(a0[1], b1, c0);  c1 = WMMA16(a1[1], b1, c1);
    c0 = WMMA16(a0[2], b2, c0);  c1 = WMMA16(a1[2], b2, c1);
    c0 = WMMA16(a0[3], b3, c0);  c1 = WMMA16(a1[3], b3, c1);

    // ---- exp2 of pre-scaled cosine + label mask, accumulated per lane ----
#pragma unroll
    for (int r = 0; r < 8; ++r) {
      const float e0 = __builtin_amdgcn_exp2f(c0[r]);
      const float e1 = __builtin_amdgcn_exp2f(c1[r]);
      aAll0[r] += e0;
      aPos0[r] += (rlab0[r] == clab) ? e0 : 0.0f;
      aAll1[r] += e1;
      aPos1[r] += (rlab1[r] == clab) ? e1 : 0.0f;
    }

    b0 = nb0; b1 = nb1; b2 = nb2; b3 = nb3; clab = nclab;
  }

  // ---- butterfly reduce across the 16 lanes of each half (column sum) ----
#pragma unroll
  for (int m = 1; m < 16; m <<= 1) {
#pragma unroll
    for (int r = 0; r < 8; ++r) {
      aAll0[r] += __shfl_xor(aAll0[r], m, 32);
      aPos0[r] += __shfl_xor(aPos0[r], m, 32);
      aAll1[r] += __shfl_xor(aAll1[r], m, 32);
      aPos1[r] += __shfl_xor(aPos1[r], m, 32);
    }
  }

  if (lm == 0) {
#pragma unroll
    for (int r = 0; r < 8; ++r) {
      atomicAdd(&sAll[hi * 8 + r],      aAll0[r]);   // ds_add_f32
      atomicAdd(&sPos[hi * 8 + r],      aPos0[r]);
      atomicAdd(&sAll[16 + hi * 8 + r], aAll1[r]);
      atomicAdd(&sPos[16 + hi * 8 + r], aPos1[r]);
    }
  }
  __syncthreads();

  if (tid < 32) {
    atomicAdd(&rowAll[row0 + tid], sAll[tid]);   // global_atomic_add_f32
    atomicAdd(&rowPos[row0 + tid], sPos[tid]);
  }
}

// ---------------------------------------------------------------------------
// Kernel 3: n_pos via class histogram, per-row loss, mean. Single block.
//   loss_i = log(all_i - e^10) - log(pos_i - e^10) + log(n_pos_i)
// ---------------------------------------------------------------------------
__global__ __launch_bounds__(256) void k_finalize(
    const int* __restrict__ labels, const float* __restrict__ rowAll,
    const float* __restrict__ rowPos, float* __restrict__ out) {
  __shared__ int   cnt[128];
  __shared__ float red[256];
  const int tid = threadIdx.x;

  if (tid < 128) cnt[tid] = 0;
  __syncthreads();
  for (int i = tid; i < NROWS; i += 256) atomicAdd(&cnt[labels[i] & 127], 1);
  __syncthreads();

  float acc = 0.0f;
  for (int i = tid; i < NROWS; i += 256) {
    const float allv = rowAll[i] - (float)EXP_SELF;
    const float posv = rowPos[i] - (float)EXP_SELF;
    const float npos = (float)(cnt[labels[i] & 127] - 1);
    acc += __logf(allv) - __logf(posv) + __logf(npos);
  }
  red[tid] = acc;
  __syncthreads();
#pragma unroll
  for (int s = 128; s > 0; s >>= 1) {
    if (tid < s) red[tid] += red[tid + s];
    __syncthreads();
  }
  if (tid == 0) out[0] = red[0] / (float)NROWS;
}

// ---------------------------------------------------------------------------
extern "C" void kernel_launch(void* const* d_in, const int* in_sizes, int n_in,
                              void* d_out, int out_size, void* d_ws, size_t ws_size,
                              hipStream_t stream) {
  (void)in_sizes; (void)n_in; (void)out_size; (void)ws_size;
  const float* feat   = (const float*)d_in[0];
  const int*   labels = (const int*)d_in[1];
  float*       out    = (float*)d_out;

  // workspace: [ fnorm f16 8192x128 | rowAll f32 8192 | rowPos f32 8192 ]
  half_t* fnorm  = (half_t*)d_ws;
  float*  rowAll = (float*)((char*)d_ws + (size_t)NROWS * KDIM * sizeof(half_t));
  float*  rowPos = rowAll + NROWS;

  k_normalize<<<NROWS / 8, 256, 0, stream>>>(feat, fnorm, rowAll, rowPos);
  k_cosexp<<<dim3(NROWS / 32, 8), 256, 0, stream>>>(fnorm, labels, rowAll, rowPos);
  k_finalize<<<1, 256, 0, stream>>>(labels, rowAll, rowPos, out);
}
